// FFN_LABEL_WISE_78254304133691
// MI455X (gfx1250) — matrix operand order repaired and checked by reading, not compiled, not run on hardware
//
#include <hip/hip_runtime.h>

#define B_ 8192
#define D_ 1024
#define L_ 64
#define H_ 256

// LDS row stride in halfwords: 32 bf16 payload + 8 pad -> 80 bytes = 5*16B (odd
// multiple of 16B => conflict-free b128 accesses across 16 lanes).
#define XSTR 40

// Per-buffer LDS layout (bytes):
//   sXh @ 0      : 128*80 = 10240
//   sXl @ 10240  : 10240
//   sWh @ 20480  : 256*80 = 20480
//   sWl @ 40960  : 20480
#define OFF_XH 0
#define OFF_XL 10240
#define OFF_WH 20480
#define OFF_WL 40960
#define BUFSZ  61440

typedef unsigned int u32;
typedef __attribute__((ext_vector_type(16))) __bf16 v16bf;
typedef __attribute__((ext_vector_type(8)))  float  v8f;
typedef __attribute__((ext_vector_type(4)))  u32    u32x4;
typedef __attribute__((ext_vector_type(8)))  int    i32x8;
typedef __attribute__((ext_vector_type(4)))  int    i32x4;

union Frag {
  uint4 u[2];
  v16bf v;
};

__device__ __forceinline__ unsigned short f2bf_rne(float f) {
  unsigned u = __float_as_uint(f);
  u += 0x7FFFu + ((u >> 16) & 1u);
  return (unsigned short)(u >> 16);
}
__device__ __forceinline__ float bf2f(unsigned short h) {
  return __uint_as_float(((unsigned)h) << 16);
}

// x [B,D] f32 -> hi/lo bf16 (same layout)
__global__ void cvt_x_kernel(const float* __restrict__ x,
                             unsigned short* __restrict__ xh,
                             unsigned short* __restrict__ xl) {
  size_t i = (size_t)blockIdx.x * blockDim.x + threadIdx.x;
  float f = x[i];
  unsigned short hi = f2bf_rne(f);
  float lo = f - bf2f(hi);
  xh[i] = hi;
  xl[i] = f2bf_rne(lo);
}

// W1 [L,D,H] f32 -> transposed hi/lo bf16 [L,H,D]
__global__ void cvt_w_kernel(const float* __restrict__ w1,
                             unsigned short* __restrict__ wh,
                             unsigned short* __restrict__ wl) {
  size_t i = (size_t)blockIdx.x * blockDim.x + threadIdx.x;  // output index
  int d = (int)(i % D_);
  int h = (int)((i / D_) % H_);
  int l = (int)(i / ((size_t)D_ * H_));
  float f = w1[((size_t)l * D_ + d) * H_ + h];
  unsigned short hi = f2bf_rne(f);
  float lo = f - bf2f(hi);
  wh[i] = hi;
  wl[i] = f2bf_rne(lo);
}

// Issue one 2-D TDM tile load: [tile_d1 rows x tile_d0 elems] of 2-byte data,
// row stride `stride0` elems in memory, written to LDS with 16B padding after
// every 64B row (pad_interval=3: 16 DWORDs, pad_amount=3: 4 DWORDs) so rows
// land at the 80-byte XSTR stride.
__device__ __forceinline__ void tdm_load_2d(u32 lds_addr, const void* gptr,
                                            u32 tile_d0, u32 tile_d1, u32 stride0) {
  unsigned long long ga = (unsigned long long)gptr;
  u32x4 g0;
  g0[0] = 1u;                                   // count=1, user descriptor
  g0[1] = lds_addr;                             // bits [63:32]
  g0[2] = (u32)ga;                              // global_addr [95:64]
  g0[3] = ((u32)(ga >> 32) & 0x01FFFFFFu) | (2u << 30);  // addr[56:32] | type=2
  i32x8 g1;
  const u32 td0 = 0x100000u, td1 = 0x100000u;   // oversized tensor dims (no OOB clamp)
  g1[0] = (int)((1u << 16) | (1u << 20) | (3u << 22) | (3u << 25));
  //           data_size=2B   pad_en      pad_int=16dw  pad_amt=4dw
  g1[1] = (int)((td0 & 0xFFFFu) << 16);                         // tensor_dim0 lo
  g1[2] = (int)(((td0 >> 16) & 0xFFFFu) | ((td1 & 0xFFFFu) << 16));
  g1[3] = (int)(((td1 >> 16) & 0xFFFFu) | (tile_d0 << 16));     // tile_dim0
  g1[4] = (int)(tile_d1 & 0xFFFFu);                             // tile_dim1 (tile_dim2=0)
  g1[5] = (int)stride0;                                         // tensor_dim0_stride lo
  g1[6] = 0;
  g1[7] = 0;
  i32x4 z4 = (i32x4)0;                          // groups 2/3 unused (<=2-D)
  i32x8 z8 = (i32x8)0;                          // unused trailing group
  __builtin_amdgcn_tensor_load_to_lds(g0, g1, z4, z4, z8, 0);
}

__global__ __launch_bounds__(256) void ffn_label_wmma_kernel(
    const unsigned short* __restrict__ xh, const unsigned short* __restrict__ xl,
    const unsigned short* __restrict__ wh, const unsigned short* __restrict__ wl,
    const float* __restrict__ b1, const float* __restrict__ w2,
    const float* __restrict__ b2, float* __restrict__ out) {
  extern __shared__ __align__(16) char smem[];  // 2 * BUFSZ bytes (double buffer)

  const int tid  = threadIdx.x;
  const int lane = tid & 31;
  const int wid  = tid >> 5;
  const int mhalf = wid >> 2;     // 0..1 : which 64-row half of the 128-row block
  const int nq    = wid & 3;      // 0..3 : which 64-col quarter of H=256
  const int l     = blockIdx.y;
  const int rowbase = blockIdx.x * 128;
  const size_t xoff = (size_t)rowbase * D_;
  const size_t woff = (size_t)l * H_ * D_;
  const u32 lds_base = (u32)(uintptr_t)&smem[0];

  v8f acc[4][4];
#pragma unroll
  for (int mt = 0; mt < 4; ++mt)
#pragma unroll
    for (int nt = 0; nt < 4; ++nt) acc[mt][nt] = (v8f)(0.0f);

  // One wave drives the Tensor Data Mover for all 4 tiles of a K-step.
  auto issueTDM = [&](int t, int buf) {
    const int k0 = t * 32;
    const u32 lb = lds_base + (u32)buf * BUFSZ;
    tdm_load_2d(lb + OFF_XH, &xh[xoff + k0], 32u, 128u, (u32)D_);
    tdm_load_2d(lb + OFF_XL, &xl[xoff + k0], 32u, 128u, (u32)D_);
    tdm_load_2d(lb + OFF_WH, &wh[woff + k0], 32u, 256u, (u32)D_);
    tdm_load_2d(lb + OFF_WL, &wl[woff + k0], 32u, 256u, (u32)D_);
  };

  // A fragment (16x32 bf16): lanes 0-15 row m, K 0-7 & 16-23; lanes 16-31 K 8-15 & 24-31
  auto loadFragA = [&](const unsigned short* s, int mbase) -> Frag {
    Frag f;
    int r = mbase + (lane & 15);
    int kh = (lane >> 4) << 3;  // 0 or 8 halfwords
    f.u[0] = *(const uint4*)&s[r * XSTR + kh];
    f.u[1] = *(const uint4*)&s[r * XSTR + kh + 16];
    return f;
  };
  // B fragment (32x16 bf16): lane n holds col h0+n; lanes 0-15 K 0-7 & 16-23; lanes 16-31 K 8-15 & 24-31
  auto loadFragB = [&](const unsigned short* s, int h0) -> Frag {
    Frag f;
    int r = h0 + (lane & 15);
    int kh = (lane >> 4) << 3;  // 0 or 8 halfwords
    f.u[0] = *(const uint4*)&s[r * XSTR + kh];
    f.u[1] = *(const uint4*)&s[r * XSTR + kh + 16];
    return f;
  };

  // Prologue: DMA K-step 0 into buffer 0.
  if (wid == 0) {
    issueTDM(0, 0);
    __builtin_amdgcn_s_wait_tensorcnt(0);
  }
  __syncthreads();

  for (int t = 0; t < 32; ++t) {
    const int cur = t & 1;
    // Prefetch next K-step via TDM while all 8 waves run WMMAs on this one.
    if (t < 31 && wid == 0) issueTDM(t + 1, cur ^ 1);

    const unsigned short* sXh = (const unsigned short*)(smem + (size_t)cur * BUFSZ + OFF_XH);
    const unsigned short* sXl = (const unsigned short*)(smem + (size_t)cur * BUFSZ + OFF_XL);
    const unsigned short* sWh = (const unsigned short*)(smem + (size_t)cur * BUFSZ + OFF_WH);
    const unsigned short* sWl = (const unsigned short*)(smem + (size_t)cur * BUFSZ + OFF_WL);

    Frag bh[4], bl[4];
#pragma unroll
    for (int nt = 0; nt < 4; ++nt) {
      bh[nt] = loadFragB(sWh, nq * 64 + nt * 16);
      bl[nt] = loadFragB(sWl, nq * 64 + nt * 16);
    }
#pragma unroll
    for (int mt = 0; mt < 4; ++mt) {
      Frag ah = loadFragA(sXh, mhalf * 64 + mt * 16);
      Frag al = loadFragA(sXl, mhalf * 64 + mt * 16);
#pragma unroll
      for (int nt = 0; nt < 4; ++nt) {
        // fp32-emulating split: Ah*Bh + Ah*Bl + Al*Bh, f32 accumulate
        acc[mt][nt] = __builtin_amdgcn_wmma_f32_16x16x32_bf16(
            false, al.v, false, bh[nt].v, (short)0, acc[mt][nt], false, false);
        acc[mt][nt] = __builtin_amdgcn_wmma_f32_16x16x32_bf16(
            false, ah.v, false, bl[nt].v, (short)0, acc[mt][nt], false, false);
        acc[mt][nt] = __builtin_amdgcn_wmma_f32_16x16x32_bf16(
            false, ah.v, false, bh[nt].v, (short)0, acc[mt][nt], false, false);
      }
    }
    // Issuing wave drains the DMA; barrier publishes the next buffer to all waves.
    if (wid == 0) __builtin_amdgcn_s_wait_tensorcnt(0);
    __syncthreads();
  }

  // ---- Fused epilogue: out[b,l] = sum_h relu(acc + b1) * W2 + b2 ----
  float* sRow = (float*)smem;  // reuse staging LDS
  if (tid < 128) sRow[tid] = 0.0f;
  __syncthreads();

  float ps[4][8];
#pragma unroll
  for (int mt = 0; mt < 4; ++mt)
#pragma unroll
    for (int r = 0; r < 8; ++r) ps[mt][r] = 0.0f;

#pragma unroll
  for (int nt = 0; nt < 4; ++nt) {
    int hcol = nq * 64 + nt * 16 + (lane & 15);
    float b1v = b1[l * H_ + hcol];
    float w2v = w2[l * H_ + hcol];
#pragma unroll
    for (int mt = 0; mt < 4; ++mt)
#pragma unroll
      for (int r = 0; r < 8; ++r) {
        float hv = acc[mt][nt][r] + b1v;
        hv = hv > 0.0f ? hv : 0.0f;
        ps[mt][r] += hv * w2v;
      }
  }

#pragma unroll
  for (int mt = 0; mt < 4; ++mt) {
#pragma unroll
    for (int r = 0; r < 8; ++r) {
      float v = ps[mt][r];
      v += __shfl_xor(v, 1, 16);
      v += __shfl_xor(v, 2, 16);
      v += __shfl_xor(v, 4, 16);
      v += __shfl_xor(v, 8, 16);
      if ((lane & 15) == 0) {
        int row = mhalf * 64 + mt * 16 + (lane >> 4) * 8 + r;
        atomicAdd(&sRow[row], v);
      }
    }
  }
  __syncthreads();

  if (tid < 128) {
    out[(size_t)(rowbase + tid) * L_ + l] = sRow[tid] + b2[l];
  }
}

extern "C" void kernel_launch(void* const* d_in, const int* in_sizes, int n_in,
                              void* d_out, int out_size, void* d_ws, size_t ws_size,
                              hipStream_t stream) {
  (void)in_sizes; (void)n_in; (void)out_size; (void)ws_size;
  const float* x  = (const float*)d_in[0];
  const float* W1 = (const float*)d_in[1];
  const float* b1 = (const float*)d_in[2];
  const float* W2 = (const float*)d_in[3];
  const float* b2 = (const float*)d_in[4];
  float* out = (float*)d_out;

  // Workspace: xh, xl [B*D] bf16 ; wh, wl [L*H*D] bf16 (transposed W1) = ~96 MB
  unsigned short* xh = (unsigned short*)d_ws;
  unsigned short* xl = xh + (size_t)B_ * D_;
  unsigned short* wh = xl + (size_t)B_ * D_;
  unsigned short* wl = wh + (size_t)L_ * H_ * D_;

  cvt_x_kernel<<<(B_ * D_) / 256, 256, 0, stream>>>(x, xh, xl);
  cvt_w_kernel<<<(L_ * H_ * D_) / 256, 256, 0, stream>>>(W1, wh, wl);

  dim3 grid(B_ / 128, L_);
  ffn_label_wmma_kernel<<<grid, dim3(256), 2 * BUFSZ, stream>>>(
      xh, xl, wh, wl, b1, W2, b2, out);
}
